// ThesisMarkowitzMinVar_63436666962537
// MI455X (gfx1250) — matrix operand order repaired and checked by reading, not compiled, not run on hardware
//
#include <hip/hip_runtime.h>

// ---------------------------------------------------------------------------
// Fused Markowitz min-variance solver for MI455X (gfx1250, wave32).
// One workgroup (256 threads = 8 waves) per batch matrix:
//   Phase 1: Q = A^T A via v_wmma_f32_16x16x32_bf16 (split-bf16, 3 products),
//            Q kept resident in LDS (256x256 f32, padded stride).
//   Phase 2: power iteration for lambda_max -> FISTA step size.
//   Phase 3: 300 FISTA iterations with capped-simplex projection
//            (60-iter bisection done with wave-local shuffles, no barriers).
// ---------------------------------------------------------------------------

#define N_ASSETS 256
#define MAXW 0.05f
#define N_ITERS 300
#define BISECT_ITERS 60
#define POWER_ITERS 30

#define QSTRIDE 260          // floats per Q row (260*4 B: 16B aligned, bank shift 4)
#define ASTRIDE 34           // shorts per transposed-A row (68 B: odd bank shift 17)
#define CHUNK 32             // K-chunk rows staged per pass

// LDS carve (bytes)
#define OFF_Q    0
#define OFF_AHI  (N_ASSETS * QSTRIDE * 4)                 // 266240
#define OFF_ALO  (OFF_AHI + N_ASSETS * ASTRIDE * 2)       // +17408
#define OFF_Y    (OFF_ALO + N_ASSETS * ASTRIDE * 2)       // +17408
#define OFF_V    (OFF_Y + N_ASSETS * 4)
#define OFF_U    (OFF_V + N_ASSETS * 4)
#define OFF_RED  (OFF_U + N_ASSETS * 4)
#define SMEM_BYTES (OFF_RED + 32 * 4)                     // 304256 <= 320KB WGP LDS

typedef __attribute__((ext_vector_type(16))) __bf16 v16bf;
typedef __attribute__((ext_vector_type(8)))  float  v8f;

union FragU {
    unsigned int u[8];
    v16bf v;
};

// Round-to-nearest-even fp32 -> bf16 (returned as raw u16)
__device__ __forceinline__ unsigned short f32_to_bf16_rne(float x) {
    unsigned int b = __float_as_uint(x);
    unsigned int r = (b + 0x7FFFu + ((b >> 16) & 1u)) >> 16;
    return (unsigned short)r;
}
__device__ __forceinline__ float bf16_bits_to_f32(unsigned short h) {
    return __uint_as_float(((unsigned int)h) << 16);
}

// Dot of Q row t (LDS, stride QSTRIDE) with x[256] (LDS), float4-vectorized.
__device__ __forceinline__ float rowdot(const float* __restrict__ Q,
                                        const float* __restrict__ x, int t) {
    const float4* qr = (const float4*)(Q + t * QSTRIDE);
    const float4* xv = (const float4*)x;
    float s = 0.0f;
#pragma unroll 8
    for (int j = 0; j < N_ASSETS / 4; ++j) {
        float4 q = qr[j];
        float4 v = xv[j];
        s += q.x * v.x + q.y * v.y + q.z * v.z + q.w * v.w;
    }
    return s;
}

// Block-wide sum, result broadcast to all 256 threads (2 barriers).
__device__ __forceinline__ float block_sum(float x, float* red, int tid) {
#pragma unroll
    for (int m = 16; m > 0; m >>= 1) x += __shfl_xor(x, m, 32);
    __syncthreads();                       // WAR guard on red[]
    if ((tid & 31) == 0) red[tid >> 5] = x;
    __syncthreads();
    float s = red[0];
#pragma unroll
    for (int w = 1; w < 8; ++w) s += red[w];
    return s;
}

// Capped-simplex projection: rows of v onto {w : sum=1, -c<=w<=c}.
// v[256] must be in LDS (barrier already done). Bisection reductions are
// wave-local (8 elems/lane, shuffles only) -> no barriers in the 60-iter loop.
__device__ __forceinline__ float project_capped(const float* __restrict__ vlds,
                                                float vt, int tid) {
    const float c = MAXW;
    const int lane = tid & 31;
    float vv[8];
#pragma unroll
    for (int q = 0; q < 8; ++q) vv[q] = vlds[lane + 32 * q];

    float mn = vv[0], mx = vv[0];
#pragma unroll
    for (int q = 1; q < 8; ++q) { mn = fminf(mn, vv[q]); mx = fmaxf(mx, vv[q]); }
#pragma unroll
    for (int m = 16; m > 0; m >>= 1) {
        mn = fminf(mn, __shfl_xor(mn, m, 32));
        mx = fmaxf(mx, __shfl_xor(mx, m, 32));
    }
    float lo = mn - c, hi = mx + c;

    for (int it = 0; it < BISECT_ITERS; ++it) {
        float mid = 0.5f * (lo + hi);
        float s = 0.0f;
#pragma unroll
        for (int q = 0; q < 8; ++q) s += fminf(fmaxf(vv[q] - mid, -c), c);
#pragma unroll
        for (int m = 16; m > 0; m >>= 1) s += __shfl_xor(s, m, 32);
        bool gt = (s > 1.0f);
        lo = gt ? mid : lo;
        hi = gt ? hi : mid;
    }
    float tau0 = 0.5f * (lo + hi);

    float cnt = 0.0f, sint = 0.0f, khi = 0.0f, klo = 0.0f;
#pragma unroll
    for (int q = 0; q < 8; ++q) {
        float z = vv[q] - tau0;
        bool in = fabsf(z) < c;
        cnt += in ? 1.0f : 0.0f;
        sint += in ? vv[q] : 0.0f;
        khi += (z >= c) ? 1.0f : 0.0f;
        klo += (z <= -c) ? 1.0f : 0.0f;
    }
#pragma unroll
    for (int m = 16; m > 0; m >>= 1) {
        cnt += __shfl_xor(cnt, m, 32);
        sint += __shfl_xor(sint, m, 32);
        khi += __shfl_xor(khi, m, 32);
        klo += __shfl_xor(klo, m, 32);
    }
    float denom = fmaxf(cnt, 1.0f);
    float tau = (sint + c * (khi - klo) - 1.0f) / denom;
    return fminf(fmaxf(vt - tau, -c), c);
}

extern "C" __global__ void __launch_bounds__(256, 1)
markowitz_minvar_kernel(const float* __restrict__ Aall, float* __restrict__ out) {
    extern __shared__ char smem[];
    float* Q = (float*)(smem + OFF_Q);
    unsigned short* AhiT = (unsigned short*)(smem + OFF_AHI);
    unsigned short* AloT = (unsigned short*)(smem + OFF_ALO);
    float* ylds = (float*)(smem + OFF_Y);
    float* vlds = (float*)(smem + OFF_V);
    float* ulds = (float*)(smem + OFF_U);
    float* red  = (float*)(smem + OFF_RED);

    const int tid = threadIdx.x;
    const int batch = blockIdx.x;
    const float* __restrict__ A = Aall + (size_t)batch * (N_ASSETS * N_ASSETS);

    const int wv = tid >> 5;
    const int lane = tid & 31;
    const int half = lane >> 4;   // which 16-lane half of the wave
    const int m16 = lane & 15;

    // ---------------- Phase 1: Q = A^T A via split-bf16 WMMA ----------------
    for (int ch = 0; ch < (N_ASSETS / CHUNK); ++ch) {
        const int i0 = ch * CHUNK;
        // Stage chunk (CHUNK x 256) transposed + split into bf16 hi/lo.
#pragma unroll 4
        for (int rep = 0; rep < CHUNK; ++rep) {
            int idx = tid + rep * 256;          // 0..8191, coalesced over j
            int i = idx >> 8;                   // row within chunk
            int j = idx & 255;                  // column of A
            float x = A[(size_t)(i0 + i) * N_ASSETS + j];
            unsigned short h = f32_to_bf16_rne(x);
            float lof = x - bf16_bits_to_f32(h);
            unsigned short l = f32_to_bf16_rne(lof);
            AhiT[j * ASTRIDE + i] = h;
            AloT[j * ASTRIDE + i] = l;
        }
        if (ch + 1 < (N_ASSETS / CHUNK)) {
            // pull next chunk toward L2/L0 while we compute this one
            __builtin_prefetch(A + (size_t)(i0 + CHUNK) * N_ASSETS + (size_t)tid * 32, 0, 1);
        }
        __syncthreads();

        // Each wave owns 32 of the 256 output 16x16 tiles.
        for (int tt = wv * 32; tt < wv * 32 + 32; ++tt) {
            const int jt = tt >> 4;   // tile row block of Q
            const int nt = tt & 15;   // tile col block of Q

            FragU a_hi, a_lo, b_hi, b_lo;
#pragma unroll
            for (int v = 0; v < 8; ++v) {
                // A-operand (M x K = 16 x 32): [m,k] = A[i0+k, jt*16+m]
                int ka = 2 * v + 8 * half + ((v >= 4) ? 8 : 0);
                // B-operand (K x N = 32 x 16): [k,n] = A[i0+k, nt*16+n]
                int kb = 2 * v + 8 * half + ((v >= 4) ? 8 : 0);
                a_hi.u[v] = *(const unsigned int*)&AhiT[(jt * 16 + m16) * ASTRIDE + ka];
                a_lo.u[v] = *(const unsigned int*)&AloT[(jt * 16 + m16) * ASTRIDE + ka];
                b_hi.u[v] = *(const unsigned int*)&AhiT[(nt * 16 + m16) * ASTRIDE + kb];
                b_lo.u[v] = *(const unsigned int*)&AloT[(nt * 16 + m16) * ASTRIDE + kb];
            }

            // C/D layout: VGPR r holds (M = r + 8*half, N = lane&15)
            float* qtile = Q + (jt * 16 + 8 * half) * QSTRIDE + nt * 16 + m16;
            v8f acc = {};
            if (ch != 0) {
#pragma unroll
                for (int r = 0; r < 8; ++r) acc[r] = qtile[r * QSTRIDE];
            }
            // Q += hi^T*hi + hi^T*lo + lo^T*hi  (split-bf16 ~= fp32)
            acc = __builtin_amdgcn_wmma_f32_16x16x32_bf16(
                false, a_hi.v, false, b_hi.v, (short)0, acc, false, false);
            acc = __builtin_amdgcn_wmma_f32_16x16x32_bf16(
                false, a_hi.v, false, b_lo.v, (short)0, acc, false, false);
            acc = __builtin_amdgcn_wmma_f32_16x16x32_bf16(
                false, a_lo.v, false, b_hi.v, (short)0, acc, false, false);
#pragma unroll
            for (int r = 0; r < 8; ++r) qtile[r * QSTRIDE] = acc[r];
        }
        __syncthreads();
    }

    // ---------------- Phase 2: power iteration -> step size ----------------
    ulds[tid] = 0.0625f;   // 1/sqrt(256)
    __syncthreads();
    for (int p = 0; p < POWER_ITERS; ++p) {
        float t = rowdot(Q, ulds, tid);
        float nrm2 = block_sum(t * t, red, tid);   // contains barriers
        float inv = 1.0f / (sqrtf(nrm2) + 1e-12f);
        ulds[tid] = t * inv;
        __syncthreads();
    }
    float qt = rowdot(Q, ulds, tid);
    float lmax = block_sum(ulds[tid] * qt, red, tid);
    const float step = 1.0f / (2.0f * lmax + 1e-12f);

    // ---------------- Phase 3: FISTA ----------------
    vlds[tid] = 1.0f / (float)N_ASSETS;
    __syncthreads();
    float w = project_capped(vlds, 1.0f / (float)N_ASSETS, tid);
    ylds[tid] = w;
    float tk = 1.0f;
    __syncthreads();

    for (int it = 0; it < N_ITERS; ++it) {
        float yt = ylds[tid];
        float g = 2.0f * rowdot(Q, ylds, tid);
        float vt = yt - step * g;
        vlds[tid] = vt;
        __syncthreads();                       // v ready; all rowdots done
        float wn = project_capped(vlds, vt, tid);
        float tn = 0.5f * (1.0f + sqrtf(1.0f + 4.0f * tk * tk));
        float yn = wn + ((tk - 1.0f) / tn) * (wn - w);
        w = wn;
        tk = tn;
        ylds[tid] = yn;
        __syncthreads();                       // y ready for next iteration
    }

    out[(size_t)batch * N_ASSETS + tid] = w;
}

extern "C" void kernel_launch(void* const* d_in, const int* in_sizes, int n_in,
                              void* d_out, int out_size, void* d_ws, size_t ws_size,
                              hipStream_t stream) {
    (void)n_in; (void)d_ws; (void)ws_size; (void)out_size;
    const float* A = (const float*)d_in[0];
    float* out = (float*)d_out;
    const int batch = in_sizes[0] / (N_ASSETS * N_ASSETS);   // 512
    markowitz_minvar_kernel<<<dim3(batch), dim3(256), SMEM_BYTES, stream>>>(A, out);
}
